// MultiHeadSelfAttention_62706522521617
// MI455X (gfx1250) — matrix operand order repaired and checked by reading, not compiled, not run on hardware
//
#include <hip/hip_runtime.h>
#include <hip/hip_bf16.h>

// ---------------------------------------------------------------------------
// MI455X (gfx1250) multi-head self-attention, bf16 WMMA pipeline.
//   B=16, S=1024, D=512, H=8, DK=DH=64
// ---------------------------------------------------------------------------

typedef __bf16 bf16;
typedef __attribute__((ext_vector_type(8)))  bf16  v8bf;
typedef __attribute__((ext_vector_type(16))) bf16  v16bf;
typedef __attribute__((ext_vector_type(8)))  float v8f;

#define BB 16
#define SS 1024
#define DD 512
#define HH 8
#define DKK 64

// ---- WMMA helpers ---------------------------------------------------------

__device__ __forceinline__ v8f wmma_bf16(v16bf a, v16bf b, v8f c) {
  // v_wmma_f32_16x16x32_bf16: (neg_a, A, neg_b, B, c_mod, C, reuse_a, reuse_b)
  return __builtin_amdgcn_wmma_f32_16x16x32_bf16(false, a, false, b, (short)0, c,
                                                 false, false);
}

// A-matrix 16x32 bf16, row-major source, leading dim `ld` (elements).
// Lane L: M = L&15, half = L>>4; elem i<8 -> K=half*8+i; i>=8 -> K=16+half*8+(i-8)
__device__ __forceinline__ v16bf load_a_tile(const bf16* __restrict__ p0, int ld) {
  const int lane = threadIdx.x & 31;
  const int m = lane & 15, half = lane >> 4;
  const bf16* p = p0 + (size_t)m * ld + half * 8;
  v8bf c0 = *(const v8bf*)(p);
  v8bf c1 = *(const v8bf*)(p + 16);
  return __builtin_shufflevector(c0, c1, 0, 1, 2, 3, 4, 5, 6, 7,
                                 8, 9, 10, 11, 12, 13, 14, 15);
}

// B-matrix 32x16 bf16 from a transposed (N-major) source: row n holds K contig.
// Lane L: N = L&15, half = L>>4; elem i<8 -> K=half*8+i; i>=8 -> K=16+half*8+(i-8)
__device__ __forceinline__ v16bf load_b_tile(const bf16* __restrict__ p0, int ld) {
  const int lane = threadIdx.x & 31;
  const int n = lane & 15, half = lane >> 4;
  const bf16* p = p0 + (size_t)n * ld + half * 8;
  v8bf c0 = *(const v8bf*)(p);
  v8bf c1 = *(const v8bf*)(p + 16);
  return __builtin_shufflevector(c0, c1, 0, 1, 2, 3, 4, 5, 6, 7,
                                 8, 9, 10, 11, 12, 13, 14, 15);
}

// ---- Conversion / layout kernels ------------------------------------------

__global__ void k_cvt_bf16(const float* __restrict__ src, bf16* __restrict__ dst,
                           int n) {
  int i = blockIdx.x * blockDim.x + threadIdx.x;
  const int stride = gridDim.x * blockDim.x;
  for (; i < n; i += stride) dst[i] = (bf16)src[i];
}

// [H, D, N(=64)] fp32 -> [H, N, D] bf16  (N-major so B-loads are contiguous in K)
__global__ void k_cvt_wT(const float* __restrict__ w, bf16* __restrict__ wT) {
  int i = blockIdx.x * blockDim.x + threadIdx.x;  // H*64*512 = 262144 exact
  const int h = i >> 15;                          // / (64*512)
  const int rem = i & 32767;
  const int n = rem >> 9;                         // / 512
  const int d = rem & 511;
  wT[i] = (bf16)w[((size_t)h * 512 + d) * 64 + n];
}

// wo [H,64,D] fp32 -> woT [D, H*64] bf16 : woT[d*512 + c] = wo[c*512 + d]
__global__ void k_cvt_woT(const float* __restrict__ wo, bf16* __restrict__ woT) {
  int i = blockIdx.x * blockDim.x + threadIdx.x;  // 512*512 = 262144 exact
  const int d = i >> 9, c = i & 511;
  woT[i] = (bf16)wo[(size_t)c * 512 + d];
}

// ---- QKV projection: [B*S,512] @ [512,64] per (head, {q,k,v}) -------------
// 1 wave -> 16x64 tile; V written transposed Vt[b][h][v][s].
__global__ void k_qkv(const bf16* __restrict__ xb, const bf16* __restrict__ wqT,
                      const bf16* __restrict__ wkT, const bf16* __restrict__ wvT,
                      bf16* __restrict__ Q, bf16* __restrict__ K,
                      bf16* __restrict__ Vt) {
  const int w = threadIdx.x >> 5;
  const int lane = threadIdx.x & 31;
  const int half = lane >> 4, nl = lane & 15;
  const int m0 = (blockIdx.x * 4 + w) * 16;  // global row in [0,16384)
  const int h = blockIdx.y;
  const int which = blockIdx.z;  // 0=Q 1=K 2=V
  const bf16* wT =
      (which == 0 ? wqT : which == 1 ? wkT : wvT) + (size_t)h * 64 * 512;

  v8f acc[4] = {v8f{}, v8f{}, v8f{}, v8f{}};
  const bf16* arow = xb + (size_t)m0 * 512;
  for (int kk = 0; kk < 512; kk += 32) {
    v16bf a = load_a_tile(arow + kk, 512);
#pragma unroll
    for (int j = 0; j < 4; ++j) {
      v16bf b = load_b_tile(wT + (size_t)(j * 16) * 512 + kk, 512);
      acc[j] = wmma_bf16(a, b, acc[j]);
    }
  }

  const int b = m0 >> 10;
  const int s0 = m0 & 1023;
  if (which < 2) {
    bf16* dst = (which == 0 ? Q : K) + (((size_t)b * HH + h) * SS + s0) * 64;
#pragma unroll
    for (int j = 0; j < 4; ++j)
#pragma unroll
      for (int r = 0; r < 8; ++r)
        dst[(size_t)(r + 8 * half) * 64 + j * 16 + nl] = (bf16)acc[j][r];
  } else {
    // transposed store: lane holds 8 consecutive s for fixed column v -> b128
#pragma unroll
    for (int j = 0; j < 4; ++j) {
      v8bf t;
#pragma unroll
      for (int r = 0; r < 8; ++r) t[r] = (bf16)acc[j][r];
      const int v = j * 16 + nl;
      *(v8bf*)(Vt + (((size_t)b * HH + h) * 64 + v) * SS + s0 + 8 * half) = t;
    }
  }
}

// ---- Flash attention: 1 wave per 16-query block ---------------------------
__global__ void k_flash(const bf16* __restrict__ Q, const bf16* __restrict__ K,
                        const bf16* __restrict__ Vt, bf16* __restrict__ ctx) {
  __shared__ __align__(16) bf16 pbuf[4][16][40];  // per-wave P transpose buffer
  const int w = threadIdx.x >> 5;
  const int lane = threadIdx.x & 31;
  const int half = lane >> 4, nl = lane & 15;
  const int task = blockIdx.x * 4 + w;  // [0, 8192)
  const int qb = task & 63;
  const int h = (task >> 6) & 7;
  const int b = task >> 9;
  const size_t bh = (size_t)b * HH + h;
  const bf16* Qb = Q + (bh * SS + qb * 16) * 64;
  const bf16* Kb = K + bh * SS * 64;
  const bf16* Vb = Vt + bh * 64 * SS;

  // Q tile stays in registers for all 32 key-steps
  const v16bf qa0 = load_a_tile(Qb + 0, 64);
  const v16bf qa1 = load_a_tile(Qb + 32, 64);

  float mrow[8], lrow[8];
#pragma unroll
  for (int r = 0; r < 8; ++r) { mrow[r] = -1e30f; lrow[r] = 0.0f; }
  v8f o[4] = {v8f{}, v8f{}, v8f{}, v8f{}};
  const float rs = 0.125f;  // 1/sqrt(64)

  for (int tb = 0; tb < SS; tb += 32) {
    v8f s0{}, s1{};
    {
      v16bf kbt;
      kbt = load_b_tile(Kb + (size_t)tb * 64 + 0, 64);         s0 = wmma_bf16(qa0, kbt, s0);
      kbt = load_b_tile(Kb + (size_t)tb * 64 + 32, 64);        s0 = wmma_bf16(qa1, kbt, s0);
      kbt = load_b_tile(Kb + (size_t)(tb + 16) * 64 + 0, 64);  s1 = wmma_bf16(qa0, kbt, s1);
      kbt = load_b_tile(Kb + (size_t)(tb + 16) * 64 + 32, 64); s1 = wmma_bf16(qa1, kbt, s1);
    }
    // online softmax on the 16x32 score tile (row M = r + 8*half, col = lane&15)
#pragma unroll
    for (int r = 0; r < 8; ++r) {
      const float a0 = s0[r] * rs;
      const float a1 = s1[r] * rs;
      float mx = fmaxf(a0, a1);
      mx = fmaxf(mx, __shfl_xor(mx, 1, 32));
      mx = fmaxf(mx, __shfl_xor(mx, 2, 32));
      mx = fmaxf(mx, __shfl_xor(mx, 4, 32));
      mx = fmaxf(mx, __shfl_xor(mx, 8, 32));
      const float mnew = fmaxf(mrow[r], mx);
      const float alpha = __expf(mrow[r] - mnew);
      const float p0 = __expf(a0 - mnew);
      const float p1 = __expf(a1 - mnew);
      float rsum = p0 + p1;
      rsum += __shfl_xor(rsum, 1, 32);
      rsum += __shfl_xor(rsum, 2, 32);
      rsum += __shfl_xor(rsum, 4, 32);
      rsum += __shfl_xor(rsum, 8, 32);
      lrow[r] = lrow[r] * alpha + rsum;
      mrow[r] = mnew;
      o[0][r] *= alpha; o[1][r] *= alpha; o[2][r] *= alpha; o[3][r] *= alpha;
      // C-layout -> LDS (row = M, col = key offset within the 32-block)
      pbuf[w][r + 8 * half][nl] = (bf16)p0;
      pbuf[w][r + 8 * half][16 + nl] = (bf16)p1;
    }
    asm volatile("s_wait_dscnt 0" ::: "memory");
    // re-read P in A-matrix layout
    const bf16* prow = &pbuf[w][nl][0];
    v8bf c0 = *(const v8bf*)(prow + half * 8);
    v8bf c1 = *(const v8bf*)(prow + 16 + half * 8);
    v16bf pa = __builtin_shufflevector(c0, c1, 0, 1, 2, 3, 4, 5, 6, 7,
                                       8, 9, 10, 11, 12, 13, 14, 15);
#pragma unroll
    for (int j = 0; j < 4; ++j) {
      v16bf vbt = load_b_tile(Vb + (size_t)(j * 16) * SS + tb, SS);
      o[j] = wmma_bf16(pa, vbt, o[j]);
    }
  }

  float inv[8];
#pragma unroll
  for (int r = 0; r < 8; ++r) inv[r] = 1.0f / lrow[r];
  // ctx layout [B, S, H*64] so the output projection is one big GEMM over K=512
  bf16* dst = ctx + ((size_t)b * SS + qb * 16) * 512 + h * 64;
#pragma unroll
  for (int j = 0; j < 4; ++j)
#pragma unroll
    for (int r = 0; r < 8; ++r)
      dst[(size_t)(r + 8 * half) * 512 + j * 16 + nl] = (bf16)(o[j][r] * inv[r]);
}

// ---- Output projection: [16384,512] @ [512,512] -> fp32 out ---------------
__global__ void k_out(const bf16* __restrict__ ctx, const bf16* __restrict__ woT,
                      float* __restrict__ out) {
  const int w = threadIdx.x >> 5;
  const int lane = threadIdx.x & 31;
  const int half = lane >> 4, nl = lane & 15;
  const int m0 = (blockIdx.x * 4 + w) * 16;
  const int n0 = blockIdx.y * 64;

  v8f acc[4] = {v8f{}, v8f{}, v8f{}, v8f{}};
  const bf16* arow = ctx + (size_t)m0 * 512;
  for (int kk = 0; kk < 512; kk += 32) {
    v16bf a = load_a_tile(arow + kk, 512);
#pragma unroll
    for (int j = 0; j < 4; ++j) {
      v16bf bt = load_b_tile(woT + (size_t)(n0 + j * 16) * 512 + kk, 512);
      acc[j] = wmma_bf16(a, bt, acc[j]);
    }
  }
#pragma unroll
  for (int j = 0; j < 4; ++j)
#pragma unroll
    for (int r = 0; r < 8; ++r)
      out[(size_t)(m0 + r + 8 * half) * 512 + n0 + j * 16 + nl] = acc[j][r];
}

// ---------------------------------------------------------------------------

extern "C" void kernel_launch(void* const* d_in, const int* in_sizes, int n_in,
                              void* d_out, int out_size, void* d_ws, size_t ws_size,
                              hipStream_t stream) {
  const float* x = (const float*)d_in[0];   // [16,1024,512]
  const float* wq = (const float*)d_in[1];  // [8,512,64]
  const float* wk = (const float*)d_in[2];  // [8,512,64]
  const float* wv = (const float*)d_in[3];  // [8,512,64]
  const float* wo = (const float*)d_in[4];  // [8,64,512]
  float* out = (float*)d_out;               // [16,1024,512]

  const size_t NX = (size_t)BB * SS * DD;  // 8388608
  const size_t NW = (size_t)HH * 64 * DD;  // 262144

  bf16* p = (bf16*)d_ws;
  bf16* xb = p;  p += NX;
  bf16* wqT = p; p += NW;
  bf16* wkT = p; p += NW;
  bf16* wvT = p; p += NW;
  bf16* woT = p; p += NW;
  bf16* Qw = p;  p += NX;
  bf16* Kw = p;  p += NX;
  bf16* Vt = p;  p += NX;
  bf16* ctx = p; p += NX;

  // 1) conversions / weight transposes
  k_cvt_bf16<<<4096, 256, 0, stream>>>(x, xb, (int)NX);
  k_cvt_wT<<<1024, 256, 0, stream>>>(wq, wqT);
  k_cvt_wT<<<1024, 256, 0, stream>>>(wk, wkT);
  k_cvt_wT<<<1024, 256, 0, stream>>>(wv, wvT);
  k_cvt_woT<<<1024, 256, 0, stream>>>(wo, woT);

  // 2) QKV projections (V stored transposed)
  k_qkv<<<dim3(256, HH, 3), 128, 0, stream>>>(xb, wqT, wkT, wvT, Qw, Kw, Vt);

  // 3) flash attention -> ctx [B,S,H*64] bf16
  k_flash<<<2048, 128, 0, stream>>>(Qw, Kw, Vt, ctx);

  // 4) output projection (+head sum via K dim) -> fp32 out
  k_out<<<dim3(256, 8), 128, 0, stream>>>(ctx, woT, out);
}
